// BiDAF_90563680404043
// MI455X (gfx1250) — matrix operand order repaired and checked by reading, not compiled, not run on hardware
//
#include <hip/hip_runtime.h>
#include <hip/hip_bf16.h>

// ---------------- CDNA5 WMMA types / helpers ----------------
typedef __attribute__((ext_vector_type(16))) __bf16 v16bf;
typedef __attribute__((ext_vector_type(8)))  float  v8f;
typedef __attribute__((ext_vector_type(4)))  unsigned int uint4v;

// 16 bf16 = two 16-byte chunks; chunks give this code's own contiguous k order (same for A and B):
//  A operand: e0..15 = k(lh*16..lh*16+15)
//  B operand: e0..15 = k(lh*16..lh*16+15)
union BfVec { v16bf v; uint4v q[2]; unsigned short u[16]; };

__device__ __forceinline__ unsigned short f2bf(float f) {
  union { float f; unsigned u; } x; x.f = f;
  unsigned u = x.u;
  return (unsigned short)((u + 0x7FFFu + ((u >> 16) & 1u)) >> 16);  // RNE
}
__device__ __forceinline__ float sigmoidf_(float x) { return 1.0f / (1.0f + __expf(-x)); }
__device__ __forceinline__ float fast_tanhf(float x) {
#if __has_builtin(__builtin_amdgcn_tanhf)
  return __builtin_amdgcn_tanhf(x);
#elif __has_builtin(__builtin_amdgcn_tanh_f32)
  return __builtin_amdgcn_tanh_f32(x);
#else
  return tanhf(x);
#endif
}

#define LB 32    // batch
#define LHID 256 // hidden
#define LG 1024  // 4*hidden
#define EMBP 320 // 300 padded to mult of 32
#define XGLS_OFF 180224  // LDS byte offset of async-staged xg tile (32x1024 f32)

// async global->LDS fetch of one thread's 128B slice of xg row t (8 x b128, ASYNCcnt)
__device__ __forceinline__ void async_fetch_xg(const float* __restrict__ xg, int t, int T,
                                               unsigned lds_addr, int m, int part) {
  const float* g = xg + ((size_t)m * T + t) * 1024 + part * 32;
#if defined(__gfx1250__)
  asm volatile(
      "global_load_async_to_lds_b128 %0, %1, off\n\t"
      "global_load_async_to_lds_b128 %0, %1, off offset:16\n\t"
      "global_load_async_to_lds_b128 %0, %1, off offset:32\n\t"
      "global_load_async_to_lds_b128 %0, %1, off offset:48\n\t"
      "global_load_async_to_lds_b128 %0, %1, off offset:64\n\t"
      "global_load_async_to_lds_b128 %0, %1, off offset:80\n\t"
      "global_load_async_to_lds_b128 %0, %1, off offset:96\n\t"
      "global_load_async_to_lds_b128 %0, %1, off offset:112"
      :: "v"(lds_addr), "v"(g) : "memory");
#else
  (void)g; (void)lds_addr;
#endif
}
__device__ __forceinline__ void wait_async0() {
#if defined(__gfx1250__)
  asm volatile("s_wait_asynccnt 0x0" ::: "memory");
#endif
}

// ---------------- embedding gather -> bf16, K padded 300->320 ----------------
__global__ void embed_bf_kernel(const int* __restrict__ idx, const float* __restrict__ wv,
                                unsigned short* __restrict__ out, int n_tok) {
  int i = blockIdx.x * blockDim.x + threadIdx.x;
  if (i >= n_tok * EMBP) return;
  int tok = i / EMBP, col = i - tok * EMBP;
  out[i] = (col < 300) ? f2bf(wv[(size_t)idx[tok] * 300 + col]) : (unsigned short)0;
}

// ---------------- W (N,K) f32 -> bf16 (N,Kp) zero-padded ----------------
__global__ void padw_bf_kernel(const float* __restrict__ W, unsigned short* __restrict__ Wb,
                               int N, int K, int Kp) {
  int i = blockIdx.x * blockDim.x + threadIdx.x;
  if (i >= N * Kp) return;
  int n = i / Kp, k = i - n * Kp;
  Wb[i] = (k < K) ? f2bf(W[(size_t)n * K + k]) : (unsigned short)0;
}

// ---------------- plain f32 -> bf16 ----------------
__global__ void tobf16_kernel(const float* __restrict__ src, unsigned short* __restrict__ dst, int n) {
  int i = blockIdx.x * blockDim.x + threadIdx.x;
  if (i < n) dst[i] = f2bf(src[i]);
}

// ---------------- Hs[r,d] = bf16(H[r,d] * w_hu[d])  (similarity A operand) ----------------
__global__ void hscale_bf_kernel(const float* __restrict__ H, const float* __restrict__ w_hu,
                                 unsigned short* __restrict__ Hs) {
  size_t i = (size_t)blockIdx.x * blockDim.x + threadIdx.x;   // 32*512*512
  if (i >= (size_t)LB * 512 * 512) return;
  int d = (int)(i & 511);
  Hs[i] = f2bf(H[i] * w_hu[d]);
}

// ---------------- Y[M,N] = Xbf[M,K] @ Wbf[N,K]^T + b0[n] + b1[n]  (WMMA bf16) -------------
// M mult of 16, K mult of 32, N mult of 16. grid(ceil(N/128), M/16), block 256.
// Inner loop: 2x b128 A-load + 2x b128 B-load + 1 wmma. No branches, no conversion.
__global__ void gemm_bf_wmma(const unsigned short* __restrict__ X, const unsigned short* __restrict__ W,
                             const float* __restrict__ b0, const float* __restrict__ b1,
                             float* __restrict__ Y, int M, int N, int K) {
  const int lane = threadIdx.x & 31, wv = threadIdx.x >> 5;
  const int nt = blockIdx.x * 8 + wv;
  const int mt = blockIdx.y;
  if (nt * 16 >= N) return;
  const int lh = lane >> 4, l16 = lane & 15;
  const unsigned short* Arow = X + (size_t)(mt * 16 + l16) * K;
  const unsigned short* Brow = W + (size_t)(nt * 16 + l16) * K;
  const int nB = nt * 16 + l16;

  v8f acc = {};
  for (int k0 = 0; k0 < K; k0 += 32) {
    BfVec a, b;
    a.q[0] = *(const uint4v*)(Arow + k0 + lh * 16);
    a.q[1] = *(const uint4v*)(Arow + k0 + lh * 16 + 8);
    b.q[0] = *(const uint4v*)(Brow + k0 + lh * 16);
    b.q[1] = *(const uint4v*)(Brow + k0 + lh * 16 + 8);
    acc = __builtin_amdgcn_wmma_f32_16x16x32_bf16(false, a.v, false, b.v, (short)0, acc, false, false);
  }
  float bias = 0.0f;
  if (b0) bias += b0[nB];
  if (b1) bias += b1[nB];
#pragma unroll
  for (int r = 0; r < 8; ++r) {            // C/D: m = r + 8*lh
    int m = mt * 16 + r + lh * 8;
    Y[(size_t)m * N + nB] = acc[r] + bias;
  }
}

// ---------------- bidirectional LSTM scan (2 workgroups: dir0 fwd, dir1 bwd) -------------
// h bf16 + c f32 LDS-resident; per-step 32x1024x256 GEMM via WMMA (ds_load_b128 A operand).
// xg[t] is staged into LDS with global_load_async_to_lds_b128 one step ahead, hiding its
// latency behind the cell update.
__global__ void __launch_bounds__(1024)
bilstm_scan(const float* __restrict__ xgF, const float* __restrict__ xgB,
            const unsigned short* __restrict__ WhhF, const unsigned short* __restrict__ WhhB,
            float* __restrict__ outF, unsigned short* __restrict__ outB,
            int T, int ostride) {
  extern __shared__ char smem[];
  float* gates = (float*)smem;                                    // LB*LG f32   (128KB)
  unsigned short* hbf = (unsigned short*)(smem + LB * LG * 4);    // LB*LHID bf16 (16KB)
  float* cst = (float*)(smem + LB * LG * 4 + LB * LHID * 2);      // LB*LHID f32  (32KB)
  float* xgls = (float*)(smem + XGLS_OFF);                        // LB*LG f32   (128KB)

  const int dir = blockIdx.x;
  const float* xg = dir ? xgB : xgF;
  const unsigned short* Whh = dir ? WhhB : WhhF;
  const int coff = dir * LHID;

  const int tix = threadIdx.x;
  const int lane = tix & 31, wv = tix >> 5;
  const int lh = lane >> 4, l16 = lane & 15;

  // this thread's async-prefetch slice: row m (batch), 128B chunk `part`
  const int m_pf = tix >> 5, part_pf = tix & 31;
  const unsigned lds_pf = XGLS_OFF + m_pf * 4096 + part_pf * 128;

  for (int e = tix; e < LB * LHID; e += 1024) { hbf[e] = 0; cst[e] = 0.0f; }
  __syncthreads();

  async_fetch_xg(xg, dir ? (T - 1) : 0, T, lds_pf, m_pf, part_pf);

  for (int s = 0; s < T; ++s) {
    const int t = dir ? (T - 1 - s) : s;
    wait_async0();        // this step's xg tile landed in LDS
    __syncthreads();      // ...and is visible to all waves (also orders hbf from prev cell)
    // --- recurrent GEMM: 128 tiles (2 Mtiles x 64 Ntiles), 4 per wave ---
#pragma unroll
    for (int q = 0; q < 4; ++q) {
      const int tid = wv * 4 + q;
      const int mt = tid & 1, nt = tid >> 1;
      const unsigned short* Arow = hbf + (mt * 16 + l16) * LHID;
      const unsigned short* Brow = Whh + (size_t)(nt * 16 + l16) * LHID;
      const int nB = nt * 16 + l16;
      v8f acc = {};
#pragma unroll
      for (int k0 = 0; k0 < LHID; k0 += 32) {
        BfVec a, b;
        a.q[0] = *(const uint4v*)(Arow + k0 + lh * 16);
        a.q[1] = *(const uint4v*)(Arow + k0 + lh * 16 + 8);
        b.q[0] = *(const uint4v*)(Brow + k0 + lh * 16);
        b.q[1] = *(const uint4v*)(Brow + k0 + lh * 16 + 8);
        acc = __builtin_amdgcn_wmma_f32_16x16x32_bf16(false, a.v, false, b.v, (short)0, acc, false, false);
      }
#pragma unroll
      for (int r = 0; r < 8; ++r) {
        int m = mt * 16 + r + lh * 8;      // batch index
        gates[m * LG + nB] = acc[r] + xgls[m * LG + nB];
      }
    }
    __syncthreads();      // gates ready; xgls fully consumed
    if (s + 1 < T) async_fetch_xg(xg, dir ? (T - 2 - s) : (s + 1), T, lds_pf, m_pf, part_pf);
    // --- cell update: 8192 (b, j) elements, 8 per thread ---
#pragma unroll
    for (int s2 = 0; s2 < 8; ++s2) {
      int e = tix * 8 + s2;
      int b = e >> 8, j = e & 255;
      float gi = gates[b * LG + j];
      float gf = gates[b * LG + 256 + j];
      float gg = gates[b * LG + 512 + j];
      float go = gates[b * LG + 768 + j];
      float c = sigmoidf_(gf) * cst[e] + sigmoidf_(gi) * fast_tanhf(gg);
      float h = sigmoidf_(go) * fast_tanhf(c);
      cst[e] = c;
      hbf[e] = f2bf(h);
      size_t oidx = ((size_t)b * T + t) * ostride + coff + j;
      if (outF) outF[oidx] = h;
      if (outB) outB[oidx] = f2bf(h);
    }
  }
}

// ---------------- row-dot: out[r] = X[r,:] . w ----------------
__global__ void rowdot_kernel(const float* __restrict__ X, const float* __restrict__ w,
                              float* __restrict__ out, int R, int D) {
  int r = blockIdx.x * (blockDim.x >> 5) + (threadIdx.x >> 5);
  if (r >= R) return;
  int lane = threadIdx.x & 31;
  float s = 0.0f;
  for (int d = lane; d < D; d += 32) s += X[(size_t)r * D + d] * w[d];
  for (int off = 16; off; off >>= 1) s += __shfl_xor(s, off, 32);
  if (lane == 0) out[r] = s;
}

// ---------------- similarity: S[b,i,j] = Hs[b,i,:] . U[b,j,:] + Hw + Uw + b_sim -----------
__global__ void sim_wmma(const unsigned short* __restrict__ Hs, const unsigned short* __restrict__ Ub16,
                         const float* __restrict__ Hw, const float* __restrict__ Uw,
                         const float* __restrict__ bsim, float* __restrict__ S) {
  const int b = blockIdx.y;
  const int lane = threadIdx.x & 31, wv = threadIdx.x >> 5;
  const int tid = blockIdx.x * 8 + wv;
  const int mt = tid >> 2, nt = tid & 3;
  const int lh = lane >> 4, l16 = lane & 15;
  const unsigned short* Arow = Hs + ((size_t)b * 512 + mt * 16 + l16) * 512;
  const unsigned short* Brow = Ub16 + ((size_t)b * 64 + nt * 16 + l16) * 512;
  const int jB = nt * 16 + l16;
  v8f acc = {};
  for (int k0 = 0; k0 < 512; k0 += 32) {
    BfVec a, bm;
    a.q[0] = *(const uint4v*)(Arow + k0 + lh * 16);
    a.q[1] = *(const uint4v*)(Arow + k0 + lh * 16 + 8);
    bm.q[0] = *(const uint4v*)(Brow + k0 + lh * 16);
    bm.q[1] = *(const uint4v*)(Brow + k0 + lh * 16 + 8);
    acc = __builtin_amdgcn_wmma_f32_16x16x32_bf16(false, a.v, false, bm.v, (short)0, acc, false, false);
  }
  float addj = Uw[b * 64 + jB] + bsim[0];
#pragma unroll
  for (int r = 0; r < 8; ++r) {
    int i = mt * 16 + r + lh * 8;
    S[((size_t)b * 512 + i) * 64 + jB] = acc[r] + Hw[b * 512 + i] + addj;
  }
}

// ---------------- softmax over j (64); writes a (f32 + bf16) and raw row max --------------
__global__ void softmax_j_kernel(float* __restrict__ S, unsigned short* __restrict__ Abf,
                                 float* __restrict__ rowmax) {
  int r = blockIdx.x * (blockDim.x >> 5) + (threadIdx.x >> 5);
  if (r >= LB * 512) return;
  int lane = threadIdx.x & 31;
  float v0 = S[(size_t)r * 64 + lane];
  float v1 = S[(size_t)r * 64 + 32 + lane];
  float m = fmaxf(v0, v1);
  for (int off = 16; off; off >>= 1) m = fmaxf(m, __shfl_xor(m, off, 32));
  float e0 = __expf(v0 - m), e1 = __expf(v1 - m);
  float ssum = e0 + e1;
  for (int off = 16; off; off >>= 1) ssum += __shfl_xor(ssum, off, 32);
  float inv = 1.0f / ssum;
  float a0 = e0 * inv, a1 = e1 * inv;
  S[(size_t)r * 64 + lane] = a0;
  S[(size_t)r * 64 + 32 + lane] = a1;
  Abf[(size_t)r * 64 + lane] = f2bf(a0);
  Abf[(size_t)r * 64 + 32 + lane] = f2bf(a1);
  if (lane == 0) rowmax[r] = m;
}

// ---------------- softmax over i (512) per batch ----------------
__global__ void softmax_i_kernel(const float* __restrict__ rowmax, float* __restrict__ bb) {
  __shared__ float red[16];
  __shared__ float stat[2];
  const int b = blockIdx.x, tix = threadIdx.x;   // 512 threads
  const int lane = tix & 31, wv = tix >> 5;
  float v = rowmax[b * 512 + tix];
  float m = v;
  for (int off = 16; off; off >>= 1) m = fmaxf(m, __shfl_xor(m, off, 32));
  if (lane == 0) red[wv] = m;
  __syncthreads();
  if (tix == 0) { float mm = red[0]; for (int i = 1; i < 16; ++i) mm = fmaxf(mm, red[i]); stat[0] = mm; }
  __syncthreads();
  float e = __expf(v - stat[0]);
  float s = e;
  for (int off = 16; off; off >>= 1) s += __shfl_xor(s, off, 32);
  if (lane == 0) red[wv] = s;
  __syncthreads();
  if (tix == 0) { float ss = 0.0f; for (int i = 0; i < 16; ++i) ss += red[i]; stat[1] = ss; }
  __syncthreads();
  bb[b * 512 + tix] = e / stat[1];
}

// ---------------- h_t[b,d] = sum_i bb[b,i] * H[b,i,d] ----------------
__global__ void ht_kernel(const float* __restrict__ bbv, const float* __restrict__ H,
                          float* __restrict__ ht) {
  const int b = blockIdx.y;
  const int d = blockIdx.x * blockDim.x + threadIdx.x;   // 512
  float s = 0.0f;
  for (int i = 0; i < 512; ++i) s += bbv[b * 512 + i] * H[((size_t)b * 512 + i) * 512 + d];
  ht[b * 512 + d] = s;
}

// ---------------- U_t[b,i,d] = sum_j a[b,i,j] * U[b,j,d]  (K=64, WMMA) --------------------
__global__ void ut_wmma(const unsigned short* __restrict__ Abf, const unsigned short* __restrict__ Ub16,
                        float* __restrict__ Ut) {
  const int b = blockIdx.y;
  const int lane = threadIdx.x & 31, wv = threadIdx.x >> 5;
  const int tid = blockIdx.x * 8 + wv;
  const int mt = tid >> 5, nt = tid & 31;
  const int lh = lane >> 4, l16 = lane & 15;
  const unsigned short* Arow = Abf + ((size_t)b * 512 + mt * 16 + l16) * 64;
  const unsigned short* Ub = Ub16 + (size_t)b * 64 * 512;
  const int nB = nt * 16 + l16;
  v8f acc = {};
#pragma unroll
  for (int k0 = 0; k0 < 64; k0 += 32) {
    BfVec a, bm;
    a.q[0] = *(const uint4v*)(Arow + k0 + lh * 16);
    a.q[1] = *(const uint4v*)(Arow + k0 + lh * 16 + 8);
#pragma unroll
    for (int e = 0; e < 16; ++e) {   // B is U^T along k: strided u16 gather
      int k = k0 + (lh << 4) + e;
      bm.u[e] = Ub[(size_t)k * 512 + nB];
    }
    acc = __builtin_amdgcn_wmma_f32_16x16x32_bf16(false, a.v, false, bm.v, (short)0, acc, false, false);
  }
#pragma unroll
  for (int r = 0; r < 8; ++r) {
    int i = mt * 16 + r + lh * 8;
    Ut[((size_t)b * 512 + i) * 512 + nB] = acc[r];
  }
}

// ---------------- G = [H, U_t, H*U_t, H*h_t]  (f32 + bf16 copies) ----------------
__global__ void gcat_kernel(const float* __restrict__ H, const float* __restrict__ Ut,
                            const float* __restrict__ ht, float* __restrict__ G,
                            unsigned short* __restrict__ Gbf) {
  size_t i = (size_t)blockIdx.x * blockDim.x + threadIdx.x;   // over 32*512*512
  if (i >= (size_t)LB * 512 * 512) return;
  int d = (int)(i & 511);
  size_t row = i >> 9;              // b*512 + t
  int b = (int)(row >> 9);
  float h = H[i], u = Ut[i];
  float hu = h * u;
  float hh = h * ht[b * 512 + d];
  float* g = G + row * 2048;
  unsigned short* gb = Gbf + row * 2048;
  g[d] = h;            gb[d] = f2bf(h);
  g[512 + d] = u;      gb[512 + d] = f2bf(u);
  g[1024 + d] = hu;    gb[1024 + d] = f2bf(hu);
  g[1536 + d] = hh;    gb[1536 + d] = f2bf(hh);
}

// ---------------- p[r] = G[r,:2048].pw[:2048] + M[r,:512].pw[2048:] + pb ----------------
__global__ void outproj_kernel(const float* __restrict__ G, const float* __restrict__ Mx,
                               const float* __restrict__ pw, const float* __restrict__ pb,
                               float* __restrict__ out) {
  int r = blockIdx.x * (blockDim.x >> 5) + (threadIdx.x >> 5);
  if (r >= LB * 512) return;
  int lane = threadIdx.x & 31;
  float s = 0.0f;
  const float* g = G + (size_t)r * 2048;
  for (int d = lane; d < 2048; d += 32) s += g[d] * pw[d];
  const float* m = Mx + (size_t)r * 512;
  for (int d = lane; d < 512; d += 32) s += m[d] * pw[2048 + d];
  for (int off = 16; off; off >>= 1) s += __shfl_xor(s, off, 32);
  if (lane == 0) out[r] = s + pb[0];
}

// ============================ host orchestration ============================
extern "C" void kernel_launch(void* const* d_in, const int* in_sizes, int n_in,
                              void* d_out, int out_size, void* d_ws, size_t ws_size,
                              hipStream_t stream) {
  const int* c_word = (const int*)d_in[0];
  const int* q_word = (const int*)d_in[1];
  const float* wvec = (const float*)d_in[2];
  const float* ctxF_Wih = (const float*)d_in[3];
  const float* ctxF_Whh = (const float*)d_in[4];
  const float* ctxF_bih = (const float*)d_in[5];
  const float* ctxF_bhh = (const float*)d_in[6];
  const float* ctxB_Wih = (const float*)d_in[7];
  const float* ctxB_Whh = (const float*)d_in[8];
  const float* ctxB_bih = (const float*)d_in[9];
  const float* ctxB_bhh = (const float*)d_in[10];
  const float* w_sim = (const float*)d_in[11];
  const float* b_sim = (const float*)d_in[12];
  const float* m1F_Wih = (const float*)d_in[13];
  const float* m1F_Whh = (const float*)d_in[14];
  const float* m1F_bih = (const float*)d_in[15];
  const float* m1F_bhh = (const float*)d_in[16];
  const float* m1B_Wih = (const float*)d_in[17];
  const float* m1B_Whh = (const float*)d_in[18];
  const float* m1B_bih = (const float*)d_in[19];
  const float* m1B_bhh = (const float*)d_in[20];
  const float* m2F_Wih = (const float*)d_in[21];
  const float* m2F_Whh = (const float*)d_in[22];
  const float* m2F_bih = (const float*)d_in[23];
  const float* m2F_bhh = (const float*)d_in[24];
  const float* m2B_Wih = (const float*)d_in[25];
  const float* m2B_Whh = (const float*)d_in[26];
  const float* m2B_bih = (const float*)d_in[27];
  const float* m2B_bhh = (const float*)d_in[28];
  const float* p1w = (const float*)d_in[29];
  const float* p1b = (const float*)d_in[30];
  const float* p2w = (const float*)d_in[31];
  const float* p2b = (const float*)d_in[32];
  float* out = (float*)d_out;

  // -------- workspace layout --------
  float* ws = (float*)d_ws;
  float* Hbuf  = ws;                          // 32*512*512
  float* Ubuf  = Hbuf + 8388608;              // 32*64*512
  float* xgA   = Ubuf + 1048576;              // 32*512*1024
  float* xgB   = xgA + 16777216;              // 32*512*1024
  float* Sbuf  = xgB + 16777216;              // 32*512*64
  float* rowmx = Sbuf + 1048576;              // 32*512
  float* bbv   = rowmx + 16384;               // 32*512
  float* Hw    = bbv + 16384;                 // 32*512
  float* Uw    = Hw + 16384;                  // 32*64
  float* htb   = Uw + 2048;                   // 32*512
  float* Utb   = htb + 16384;                 // 32*512*512
  float* Gbuf  = Utb + 8388608;               // 32*512*2048
  float* M2    = Gbuf + 33554432;             // 32*512*512
  unsigned short* b16 = (unsigned short*)(M2 + 8388608);
  unsigned short* c_embbf = b16;              // 16384*320
  unsigned short* q_embbf = c_embbf + 5242880;    // 2048*320
  unsigned short* Gbf     = q_embbf + 655360;     // 16384*2048
  unsigned short* M1bf    = Gbf + 33554432;       // 16384*512
  unsigned short* Hsbf    = M1bf + 8388608;       // 16384*512
  unsigned short* Ubf     = Hsbf + 8388608;       // 2048*512
  unsigned short* Abf     = Ubf + 1048576;        // 16384*64
  unsigned short* wih_cF  = Abf + 1048576;        // 1024*320
  unsigned short* wih_cB  = wih_cF + 327680;      // 1024*320
  unsigned short* wih_1F  = wih_cB + 327680;      // 1024*2048
  unsigned short* wih_1B  = wih_1F + 2097152;     // 1024*2048
  unsigned short* wih_2F  = wih_1B + 2097152;     // 1024*512
  unsigned short* wih_2B  = wih_2F + 524288;      // 1024*512
  unsigned short* whh_cF  = wih_2B + 524288;      // 6 x 1024*256
  unsigned short* whh_cB  = whh_cF + 262144;
  unsigned short* whh_1F  = whh_cB + 262144;
  unsigned short* whh_1B  = whh_1F + 262144;
  unsigned short* whh_2F  = whh_1B + 262144;
  unsigned short* whh_2B  = whh_2F + 262144;

  const size_t scan_lds = XGLS_OFF + (size_t)LB * LG * 4;   // 304KB (<= 320KB/WGP)

  // 1) embeddings (bf16, K padded to 320)
  embed_bf_kernel<<<(16384 * EMBP + 255) / 256, 256, 0, stream>>>(c_word, wvec, c_embbf, 16384);
  embed_bf_kernel<<<(2048 * EMBP + 255) / 256, 256, 0, stream>>>(q_word, wvec, q_embbf, 2048);

  // 2) weight conversions (bf16; Wih K-padded)
  padw_bf_kernel<<<1280, 256, 0, stream>>>(ctxF_Wih, wih_cF, 1024, 300, EMBP);
  padw_bf_kernel<<<1280, 256, 0, stream>>>(ctxB_Wih, wih_cB, 1024, 300, EMBP);
  padw_bf_kernel<<<8192, 256, 0, stream>>>(m1F_Wih, wih_1F, 1024, 2048, 2048);
  padw_bf_kernel<<<8192, 256, 0, stream>>>(m1B_Wih, wih_1B, 1024, 2048, 2048);
  padw_bf_kernel<<<2048, 256, 0, stream>>>(m2F_Wih, wih_2F, 1024, 512, 512);
  padw_bf_kernel<<<2048, 256, 0, stream>>>(m2B_Wih, wih_2B, 1024, 512, 512);
  tobf16_kernel<<<1024, 256, 0, stream>>>(ctxF_Whh, whh_cF, 262144);
  tobf16_kernel<<<1024, 256, 0, stream>>>(ctxB_Whh, whh_cB, 262144);
  tobf16_kernel<<<1024, 256, 0, stream>>>(m1F_Whh, whh_1F, 262144);
  tobf16_kernel<<<1024, 256, 0, stream>>>(m1B_Whh, whh_1B, 262144);
  tobf16_kernel<<<1024, 256, 0, stream>>>(m2F_Whh, whh_2F, 262144);
  tobf16_kernel<<<1024, 256, 0, stream>>>(m2B_Whh, whh_2B, 262144);

  // 3) context BiLSTM over C (T=512)
  gemm_bf_wmma<<<dim3(8, 1024), 256, 0, stream>>>(c_embbf, wih_cF, ctxF_bih, ctxF_bhh, xgA, 16384, 1024, EMBP);
  gemm_bf_wmma<<<dim3(8, 1024), 256, 0, stream>>>(c_embbf, wih_cB, ctxB_bih, ctxB_bhh, xgB, 16384, 1024, EMBP);
  bilstm_scan<<<2, 1024, scan_lds, stream>>>(xgA, xgB, whh_cF, whh_cB, Hbuf, (unsigned short*)nullptr, 512, 512);

  // 4) context BiLSTM over Q (T=64); also emit U in bf16 for attention
  gemm_bf_wmma<<<dim3(8, 128), 256, 0, stream>>>(q_embbf, wih_cF, ctxF_bih, ctxF_bhh, xgA, 2048, 1024, EMBP);
  gemm_bf_wmma<<<dim3(8, 128), 256, 0, stream>>>(q_embbf, wih_cB, ctxB_bih, ctxB_bhh, xgB, 2048, 1024, EMBP);
  bilstm_scan<<<2, 1024, scan_lds, stream>>>(xgA, xgB, whh_cF, whh_cB, Ubuf, Ubf, 64, 512);

  // 5) attention
  rowdot_kernel<<<2048, 256, 0, stream>>>(Hbuf, w_sim, Hw, 16384, 512);
  rowdot_kernel<<<256, 256, 0, stream>>>(Ubuf, w_sim + 512, Uw, 2048, 512);
  hscale_bf_kernel<<<32768, 256, 0, stream>>>(Hbuf, w_sim + 1024, Hsbf);
  sim_wmma<<<dim3(16, 32), 256, 0, stream>>>(Hsbf, Ubf, Hw, Uw, b_sim, Sbuf);
  softmax_j_kernel<<<2048, 256, 0, stream>>>(Sbuf, Abf, rowmx);
  softmax_i_kernel<<<32, 512, 0, stream>>>(rowmx, bbv);
  ht_kernel<<<dim3(2, 32), 256, 0, stream>>>(bbv, Hbuf, htb);
  ut_wmma<<<dim3(128, 32), 256, 0, stream>>>(Abf, Ubf, Utb);
  gcat_kernel<<<32768, 256, 0, stream>>>(Hbuf, Utb, htb, Gbuf, Gbf);

  // 6) modeling BiLSTM 1 (K=2048); M1 kept bf16-only (pure GEMM input)
  gemm_bf_wmma<<<dim3(8, 1024), 256, 0, stream>>>(Gbf, wih_1F, m1F_bih, m1F_bhh, xgA, 16384, 1024, 2048);
  gemm_bf_wmma<<<dim3(8, 1024), 256, 0, stream>>>(Gbf, wih_1B, m1B_bih, m1B_bhh, xgB, 16384, 1024, 2048);
  bilstm_scan<<<2, 1024, scan_lds, stream>>>(xgA, xgB, whh_1F, whh_1B, (float*)nullptr, M1bf, 512, 512);

  // 7) modeling BiLSTM 2 (K=512)
  gemm_bf_wmma<<<dim3(8, 1024), 256, 0, stream>>>(M1bf, wih_2F, m2F_bih, m2F_bhh, xgA, 16384, 1024, 512);
  gemm_bf_wmma<<<dim3(8, 1024), 256, 0, stream>>>(M1bf, wih_2B, m2B_bih, m2B_bhh, xgB, 16384, 1024, 512);
  bilstm_scan<<<2, 1024, scan_lds, stream>>>(xgA, xgB, whh_2F, whh_2B, M2, (unsigned short*)nullptr, 512, 512);

  // 8) output projections: p1 | p2
  outproj_kernel<<<2048, 256, 0, stream>>>(Gbuf, M2, p1w, p1b, out);
  outproj_kernel<<<2048, 256, 0, stream>>>(Gbuf, M2, p2w, p2b, out + 16384);
}
